// SwinTransformer3D_v2_34205119545824
// MI455X (gfx1250) — compile-verified, matrix operand order checked by reading, not run on hardware
//
#include <hip/hip_runtime.h>
#include <hip/hip_bf16.h>

typedef __attribute__((ext_vector_type(16))) _Float16 v16h;
typedef __attribute__((ext_vector_type(8)))  _Float16 v8h;
typedef __attribute__((ext_vector_type(4)))  _Float16 v4h;
typedef __attribute__((ext_vector_type(8)))  float    v8f;
typedef __attribute__((ext_vector_type(4)))  unsigned int v4u;
typedef __attribute__((ext_vector_type(8)))  unsigned int v8u;

#define NTOK 64
#define DIM  192
#define NH   6
#define HD   32
#define NWIN 216
#define NB   1728
#define PPAD 72    // LDS row stride (halfs) for P tile

union FragU { v16h v; v8h h[2]; };

// 16x32 f16 A-fragment (or matching B-fragment for a row-major W / k / v^T row).
// ISA layout: lanes 0-15 -> row = base+lane, K = {0..7, 16..23}; lanes 16-31 -> K = {8..15, 24..31}.
__device__ __forceinline__ v16h load_frag(const _Float16* p, int row, int k0, int ld, int lane) {
  const int r  = row + (lane & 15);
  const int kh = (lane & 16) ? 8 : 0;
  const _Float16* base = p + (size_t)r * ld + k0 + kh;
  FragU f;
  f.h[0] = *(const v8h*)(base);
  f.h[1] = *(const v8h*)(base + 16);
  return f.v;
}

__device__ __forceinline__ v8f wmma_f16(v16h a, v16h b, v8f c) {
  return __builtin_amdgcn_wmma_f32_16x16x32_f16(false, a, false, b, (short)0, c, false, false);
}

// ---------------------------------------------------------------------------
// Prep: vectorized f32 -> f16 conversion (x and weights; all sizes % 4 == 0)
// ---------------------------------------------------------------------------
__global__ void k_cvt4(const float* __restrict__ src, _Float16* __restrict__ dst, int n4) {
  const int i = blockIdx.x * blockDim.x + threadIdx.x;
  if (i < n4) {
    const float4 v = ((const float4*)src)[i];
    v4h o;
    o[0] = (_Float16)v.x; o[1] = (_Float16)v.y; o[2] = (_Float16)v.z; o[3] = (_Float16)v.w;
    *(v4h*)(dst + (size_t)i * 4) = o;
  }
}

// ---------------------------------------------------------------------------
// Prep: CPB MLP -> 16*sigmoid bias table rpb[h][m][n], plus per-head scales
// ---------------------------------------------------------------------------
__global__ __launch_bounds__(512) void k_cpb(
    const float* __restrict__ w1, const float* __restrict__ b1,
    const float* __restrict__ w2, const float* __restrict__ ls,
    float* __restrict__ rpb, float* __restrict__ scales)
{
  __shared__ float rh[343 * NH];
  const int t = threadIdx.x;
  if (t < 343) {
    const int iv[3] = { t / 49, (t / 7) % 7, t % 7 };
    float tc[3];
    #pragma unroll
    for (int k = 0; k < 3; ++k) {
      const float v = (float)(iv[k] - 3) * (8.0f / 3.0f);
      const float sg = (v > 0.f) ? 1.0f : ((v < 0.f) ? -1.0f : 0.0f);
      tc[k] = sg * log2f(fabsf(v) + 1.0f) * (1.0f / 3.0f);   // /log2(8)
    }
    float o[NH] = {0.f, 0.f, 0.f, 0.f, 0.f, 0.f};
    for (int j = 0; j < 512; ++j) {
      float hsum = fmaf(tc[0], w1[j * 3 + 0],
                   fmaf(tc[1], w1[j * 3 + 1],
                   fmaf(tc[2], w1[j * 3 + 2], b1[j])));
      hsum = fmaxf(hsum, 0.0f);
      #pragma unroll
      for (int hh = 0; hh < NH; ++hh) o[hh] += hsum * w2[hh * 512 + j];
    }
    #pragma unroll
    for (int hh = 0; hh < NH; ++hh) rh[t * NH + hh] = o[hh];
  }
  if (t < NH) scales[t] = __expf(fminf(ls[t], logf(100.0f)));
  __syncthreads();
  for (int e = t; e < NH * NTOK * NTOK; e += 512) {
    const int hh = e >> 12, m = (e >> 6) & 63, n = e & 63;
    const int dm = (m >> 4) - (n >> 4);
    const int dh = ((m >> 2) & 3) - ((n >> 2) & 3);
    const int dw = (m & 3) - (n & 3);
    const int idx = ((dm + 3) * 7 + (dh + 3)) * 7 + (dw + 3);
    const float v = rh[idx * NH + hh];
    rpb[e] = 16.0f / (1.0f + __expf(-v));
  }
}

// ---------------------------------------------------------------------------
// Stage 2: qkv GEMM + cosine-normalization. One block per window, 4 waves.
// x tile arrives via TDM (tensor_load_to_lds); weight slabs (32x192 f16, 12KB)
// are staged to LDS once per block, double-buffered, prefetched into registers
// so global latency overlaps the WMMAs. v is written transposed ([hd][tok]).
// ---------------------------------------------------------------------------
__global__ __launch_bounds__(128) void k_qkv(
    const _Float16* __restrict__ xh, const _Float16* __restrict__ qkvwh,
    _Float16* __restrict__ qh, _Float16* __restrict__ kh, _Float16* __restrict__ vt)
{
  __shared__ _Float16 xs[NTOK * DIM];          // 24 KB, TDM destination
  __shared__ _Float16 wslab[2][32 * DIM];      // 2 x 12 KB double buffer
  const int b = blockIdx.x;
  const int t = threadIdx.x;
  const int wv = t >> 5, lane = t & 31;
  const int ncol = lane & 15, hi = (lane >> 4) & 1;

  // --- TDM: DMA 64x192 f16 x-tile into LDS (one issue; TENSOR op ignores EXEC)
  if (t == 0) {
    const unsigned long long ga = (unsigned long long)(const void*)(xh + (size_t)b * NTOK * DIM);
    const unsigned int lds = (unsigned int)(uintptr_t)(&xs[0]);
    v4u g0; v8u g1;
    g0[0] = 1u;                                   // count=1, user descriptor
    g0[1] = lds;                                  // lds_addr
    g0[2] = (unsigned int)ga;                     // global_addr[31:0]
    g0[3] = (unsigned int)(ga >> 32) | (2u << 30);// global_addr[56:32] | type=2
    g1[0] = 0x00010000u;                          // data_size = 2 bytes
    g1[1] = (unsigned int)DIM << 16;              // tensor_dim0 = 192 (bits 79:48)
    g1[2] = (unsigned int)NTOK << 16;             // tensor_dim1 = 64  (bits 111:80)
    g1[3] = (unsigned int)DIM << 16;              // tile_dim0 = 192   (bits 127:112)
    g1[4] = (unsigned int)NTOK;                   // tile_dim1 = 64, tile_dim2 = 0
    g1[5] = (unsigned int)DIM;                    // tensor_dim0_stride = 192
    g1[6] = 0u;
    g1[7] = 0u;
    asm volatile("tensor_load_to_lds %0, %1" :: "s"(g0), "s"(g1) : "memory");
  }

  // --- stage slab 0 while the TDM streams (slab s = W rows [(s%3)*192+(s/3)*32, +32))
  const uint4* wsrc = (const uint4*)qkvwh;    // 16B chunks; each slab = 768 chunks
  {
    const size_t base = ((size_t)0 * DIM + (size_t)0 * HD) * DIM / 8;
    uint4* dstb = (uint4*)&wslab[0][0];
    #pragma unroll
    for (int j = 0; j < 3; ++j) dstb[t + 128 * j] = wsrc[base + t + 128 * j];
  }
  if (t == 0) __builtin_amdgcn_s_wait_tensorcnt(0);
  __syncthreads();

  // cache A fragments (this wave's M tile row, all 6 K steps)
  v16h afr[6];
  #pragma unroll
  for (int ks = 0; ks < 6; ++ks) afr[ks] = load_frag(xs, 16 * wv, ks * 32, DIM, lane);

  #pragma unroll 1
  for (int s = 0; s < 18; ++s) {
    const int h = s / 3, which = s % 3;
    const _Float16* wb = &wslab[s & 1][0];

    // prefetch next slab into registers (hidden behind the 12 WMMAs below)
    uint4 pf[3];
    if (s + 1 < 18) {
      const int s2 = s + 1;
      const size_t base = ((size_t)(s2 % 3) * DIM + (size_t)(s2 / 3) * HD) * DIM / 8;
      #pragma unroll
      for (int j = 0; j < 3; ++j) pf[j] = wsrc[base + t + 128 * j];
    }

    v8f a0 = {}, a1 = {};
    #pragma unroll
    for (int ks = 0; ks < 6; ++ks) {
      const v16h bw0 = load_frag(wb, 0,  ks * 32, DIM, lane);
      const v16h bw1 = load_frag(wb, 16, ks * 32, DIM, lane);
      a0 = wmma_f16(afr[ks], bw0, a0);
      a1 = wmma_f16(afr[ks], bw1, a1);
    }

    if (which < 2) {
      // L2-normalize each token row over hd=32 (two column tiles, 16 lanes)
      #pragma unroll
      for (int r = 0; r < 8; ++r) {
        float ss = a0[r] * a0[r] + a1[r] * a1[r];
        ss += __shfl_xor(ss, 1, 32);
        ss += __shfl_xor(ss, 2, 32);
        ss += __shfl_xor(ss, 4, 32);
        ss += __shfl_xor(ss, 8, 32);
        const float inv = 1.0f / fmaxf(sqrtf(ss), 1e-12f);
        a0[r] *= inv; a1[r] *= inv;
      }
      _Float16* dst = (which == 0 ? qh : kh) + ((size_t)b * NH + h) * NTOK * HD;
      #pragma unroll
      for (int r = 0; r < 8; ++r) {
        const int m = 16 * wv + r + 8 * hi;
        dst[m * HD + ncol]      = (_Float16)a0[r];
        dst[m * HD + 16 + ncol] = (_Float16)a1[r];
      }
    } else {
      _Float16* dv = vt + ((size_t)b * NH + h) * HD * NTOK;
      v8h p0, p1;
      #pragma unroll
      for (int r = 0; r < 8; ++r) { p0[r] = (_Float16)a0[r]; p1[r] = (_Float16)a1[r]; }
      const int mb = 16 * wv + 8 * hi;   // 8 consecutive tokens per lane
      *(v8h*)(dv + ncol * NTOK + mb)        = p0;
      *(v8h*)(dv + (16 + ncol) * NTOK + mb) = p1;
    }

    if (s + 1 < 18) {
      uint4* dstb = (uint4*)&wslab[(s + 1) & 1][0];
      #pragma unroll
      for (int j = 0; j < 3; ++j) dstb[t + 128 * j] = pf[j];
    }
    __syncthreads();
  }
}

// ---------------------------------------------------------------------------
// Stage 3: attention. One block per window; wave = head. Processed one M-tile
// row at a time to halve accumulator pressure: S row (4 WMMA) -> fused
// scale+rpb+mask -> shuffle softmax -> P row to LDS; then O = P@v (16 WMMA).
// ---------------------------------------------------------------------------
__global__ __launch_bounds__(192) void k_attn(
    const _Float16* __restrict__ qh, const _Float16* __restrict__ kh,
    const _Float16* __restrict__ vt, const float* __restrict__ rpb,
    const float* __restrict__ mask, const float* __restrict__ scales,
    _Float16* __restrict__ oh)
{
  __shared__ _Float16 ph[NH * NTOK * PPAD];
  const int b = blockIdx.x;
  const int h = threadIdx.x >> 5, lane = threadIdx.x & 31;
  const int ncol = lane & 15, hi = (lane >> 4) & 1;

  const _Float16* qb = qh + ((size_t)b * NH + h) * NTOK * HD;
  const _Float16* kb = kh + ((size_t)b * NH + h) * NTOK * HD;

  v16h bk[4];
  #pragma unroll
  for (int j = 0; j < 4; ++j) bk[j] = load_frag(kb, 16 * j, 0, HD, lane);

  const float sc = scales[h];
  const float* rp = rpb + h * NTOK * NTOK;
  const float* mk = mask + (size_t)(b % NWIN) * NTOK * NTOK;
  _Float16* pb = ph + h * NTOK * PPAD;

  #pragma unroll 1
  for (int i = 0; i < 4; ++i) {
    const v16h aq = load_frag(qb, 16 * i, 0, HD, lane);
    v8f sj[4];
    #pragma unroll
    for (int j = 0; j < 4; ++j) { v8f z = {}; sj[j] = wmma_f16(aq, bk[j], z); }

    #pragma unroll
    for (int j = 0; j < 4; ++j)
      #pragma unroll
      for (int r = 0; r < 8; ++r) {
        const int m = 16 * i + r + 8 * hi, n = 16 * j + ncol;
        sj[j][r] = sj[j][r] * sc + rp[m * NTOK + n] + mk[m * NTOK + n];
      }

    // row softmax: each row's 16 columns live in one 16-lane half; masks 1..8 stay in-half
    #pragma unroll
    for (int r = 0; r < 8; ++r) {
      float mx = fmaxf(fmaxf(sj[0][r], sj[1][r]), fmaxf(sj[2][r], sj[3][r]));
      mx = fmaxf(mx, __shfl_xor(mx, 1, 32));
      mx = fmaxf(mx, __shfl_xor(mx, 2, 32));
      mx = fmaxf(mx, __shfl_xor(mx, 4, 32));
      mx = fmaxf(mx, __shfl_xor(mx, 8, 32));
      float sum = 0.f;
      #pragma unroll
      for (int j = 0; j < 4; ++j) { const float p = __expf(sj[j][r] - mx); sj[j][r] = p; sum += p; }
      sum += __shfl_xor(sum, 1, 32);
      sum += __shfl_xor(sum, 2, 32);
      sum += __shfl_xor(sum, 4, 32);
      sum += __shfl_xor(sum, 8, 32);
      const float inv = 1.0f / sum;
      #pragma unroll
      for (int j = 0; j < 4; ++j) sj[j][r] *= inv;
    }

    #pragma unroll
    for (int j = 0; j < 4; ++j)
      #pragma unroll
      for (int r = 0; r < 8; ++r) {
        const int m = 16 * i + r + 8 * hi, n = 16 * j + ncol;
        pb[m * PPAD + n] = (_Float16)sj[j][r];
      }
  }
  // same-wave DS store->load ordering is guaranteed (DScnt, in-order)

  const _Float16* vb = vt + ((size_t)b * NH + h) * HD * NTOK;
  _Float16* ob = oh + (size_t)b * NTOK * DIM;
  #pragma unroll
  for (int jt = 0; jt < 2; ++jt)
    #pragma unroll 1
    for (int i = 0; i < 4; ++i) {
      v8f o = {};
      #pragma unroll
      for (int ks = 0; ks < 2; ++ks) {
        const v16h ap = load_frag(pb, 16 * i, ks * 32, PPAD, lane);
        const v16h bv = load_frag(vb, 16 * jt, ks * 32, NTOK, lane);  // vt rows == B-frag of v
        o = wmma_f16(ap, bv, o);
      }
      #pragma unroll
      for (int r = 0; r < 8; ++r) {
        const int m = 16 * i + r + 8 * hi;
        ob[m * DIM + h * HD + 16 * jt + ncol] = (_Float16)o[r];
      }
    }
}

// ---------------------------------------------------------------------------
// Stage 4: output projection + bias. One block per window, 6 waves; each wave
// owns 2 of 12 N-tiles across all 4 M-tiles; A-fragments cached per M-tile.
// ---------------------------------------------------------------------------
__global__ __launch_bounds__(192) void k_proj(
    const _Float16* __restrict__ oh, const _Float16* __restrict__ pwh,
    const float* __restrict__ pbias, float* __restrict__ out)
{
  const int b = blockIdx.x;
  const int wv = threadIdx.x >> 5, lane = threadIdx.x & 31;
  const int ncol = lane & 15, hi = (lane >> 4) & 1;
  const _Float16* ob = oh + (size_t)b * NTOK * DIM;
  float* ub = out + (size_t)b * NTOK * DIM;

  #pragma unroll 1
  for (int i = 0; i < 4; ++i) {
    v16h af[6];
    #pragma unroll
    for (int ks = 0; ks < 6; ++ks) af[ks] = load_frag(ob, 16 * i, ks * 32, DIM, lane);
    #pragma unroll 1
    for (int jj = 0; jj < 2; ++jj) {
      const int ct = wv * 2 + jj;         // N tile 0..11
      v8f acc = {};
      #pragma unroll
      for (int ks = 0; ks < 6; ++ks) {
        const v16h bf = load_frag(pwh, ct * 16, ks * 32, DIM, lane);  // proj_w rows == B-frag of W^T
        acc = wmma_f16(af[ks], bf, acc);
      }
      const int o = ct * 16 + ncol;
      const float bias = pbias[o];
      #pragma unroll
      for (int r = 0; r < 8; ++r) {
        const int m = 16 * i + r + 8 * hi;
        ub[m * DIM + o] = acc[r] + bias;
      }
    }
  }
}

// ---------------------------------------------------------------------------
extern "C" void kernel_launch(void* const* d_in, const int* in_sizes, int n_in,
                              void* d_out, int out_size, void* d_ws, size_t ws_size,
                              hipStream_t stream) {
  (void)in_sizes; (void)n_in; (void)out_size; (void)ws_size;
  const float* x    = (const float*)d_in[0];
  const float* mask = (const float*)d_in[1];
  const float* qkvw = (const float*)d_in[2];
  const float* ls   = (const float*)d_in[3];
  const float* w1   = (const float*)d_in[4];
  const float* b1   = (const float*)d_in[5];
  const float* w2   = (const float*)d_in[6];
  const float* pw   = (const float*)d_in[7];
  const float* pb   = (const float*)d_in[8];
  float* out = (float*)d_out;

  char* ws = (char*)d_ws;
  size_t off = 0;
  auto take = [&](size_t bytes) -> char* {
    char* p = ws + off;
    off = (off + bytes + 255) & ~(size_t)255;
    return p;
  };
  _Float16* qkvwh = (_Float16*)take((size_t)3 * DIM * DIM * 2);
  _Float16* pwh   = (_Float16*)take((size_t)DIM * DIM * 2);
  float*    rpb   = (float*)   take((size_t)NH * NTOK * NTOK * 4);
  float*    scl   = (float*)   take(256);
  _Float16* xhh   = (_Float16*)take((size_t)NB * NTOK * DIM * 2);
  _Float16* qh    = (_Float16*)take((size_t)NB * NH * NTOK * HD * 2);
  _Float16* kh    = (_Float16*)take((size_t)NB * NH * NTOK * HD * 2);
  _Float16* vt    = (_Float16*)take((size_t)NB * NH * NTOK * HD * 2);
  _Float16* oh    = (_Float16*)take((size_t)NB * NTOK * DIM * 2);

  const int xn4 = NB * NTOK * DIM / 4;
  k_cvt4<<<(xn4 + 255) / 256, 256, 0, stream>>>(x, xhh, xn4);
  k_cvt4<<<(3 * DIM * DIM / 4 + 255) / 256, 256, 0, stream>>>(qkvw, qkvwh, 3 * DIM * DIM / 4);
  k_cvt4<<<(DIM * DIM / 4 + 255) / 256, 256, 0, stream>>>(pw, pwh, DIM * DIM / 4);
  k_cpb<<<1, 512, 0, stream>>>(w1, b1, w2, ls, rpb, scl);
  k_qkv<<<NB, 128, 0, stream>>>(xhh, qkvwh, qh, kh, vt);
  k_attn<<<NB, 192, 0, stream>>>(qh, kh, vt, rpb, mask, scl, oh);
  k_proj<<<NB, 192, 0, stream>>>(oh, pwh, pb, out);
}
